// LMUCell_Modified_57913339019728
// MI455X (gfx1250) — matrix operand order repaired and checked by reading, not compiled, not run on hardware
//
#include <hip/hip_runtime.h>
#include <math.h>

typedef __attribute__((ext_vector_type(16))) __bf16 v16bf;
typedef __attribute__((ext_vector_type(8)))  float  v8f;

#define UNITS   512
#define ORDER   256
#define BATCH   64
#define TSTEPS  1024
#define NWG     16
#define HCOLS   (UNITS/NWG)   // 32 hidden columns per WG
#define MCOLS   (ORDER/NWG)   // 16 memory columns per WG
#define NTHREADS 512          // 16 waves of 32

// workspace layout (bytes), all 256-aligned
#define OFF_BAR  0
#define OFF_H    1024                                 // 2 x [64][512] f32 master
#define OFF_M    (OFF_H   + 2*BATCH*UNITS*4)          // 2 x [64][256] f32 master
#define OFF_HKT  (OFF_M   + 2*BATCH*ORDER*4)          // hk^T  [512][512] bf16
#define OFF_MKT  (OFF_HKT + UNITS*UNITS*2)            // mk^T  [512][256] bf16
#define OFF_ATT  (OFF_MKT + UNITS*ORDER*2)            // AT^T  [256][256] bf16
#define OFF_HBF  (OFF_ATT + ORDER*ORDER*2)            // 2 x [64][512] bf16 mirror
#define OFF_MBF  (OFF_HBF + 2*BATCH*UNITS*2)          // 2 x [64][256] bf16 mirror

// ---------------------------------------------------------------- prep ------
__global__ void lmu_prep(const float* __restrict__ hk, const float* __restrict__ mk,
                         const float* __restrict__ AT,
                         __bf16* __restrict__ hkT, __bf16* __restrict__ mkT,
                         __bf16* __restrict__ atT,
                         float* __restrict__ hbuf, float* __restrict__ mbuf,
                         __bf16* __restrict__ hbf, __bf16* __restrict__ mbf,
                         int* __restrict__ bar) {
    const int NH = UNITS*UNITS, NM = UNITS*ORDER, NA = ORDER*ORDER;
    const int NHB = 2*BATCH*UNITS, NMB = 2*BATCH*ORDER;
    const int total = NH + NM + NA + 2*NHB + 2*NMB + 64;
    for (int i = blockIdx.x*blockDim.x + threadIdx.x; i < total;
         i += gridDim.x*blockDim.x) {
        int j = i;
        if (j < NH)  { int n = j/UNITS, k = j%UNITS; hkT[j] = (__bf16)hk[k*UNITS + n]; continue; } j -= NH;
        if (j < NM)  { int n = j/ORDER, k = j%ORDER; mkT[j] = (__bf16)mk[k*UNITS + n]; continue; } j -= NM;
        if (j < NA)  { int n = j/ORDER, k = j%ORDER; atT[j] = (__bf16)AT[k*ORDER + n]; continue; } j -= NA;
        if (j < NHB) { hbuf[j] = 0.f; continue; }             j -= NHB;
        if (j < NMB) { mbuf[j] = 0.f; continue; }             j -= NMB;
        if (j < NHB) { hbf[j] = (__bf16)0.f; continue; }      j -= NHB;
        if (j < NMB) { mbf[j] = (__bf16)0.f; continue; }      j -= NMB;
        bar[j] = 0;
    }
}

// ------------------------------------------------------------ barrier -------
__device__ __forceinline__ void grid_barrier(int* bar) {
    __syncthreads();
    if (threadIdx.x == 0) {
        __threadfence();  // release prior global stores (agent scope)
        int ticket = __hip_atomic_fetch_add(bar, 1, __ATOMIC_ACQ_REL,
                                            __HIP_MEMORY_SCOPE_AGENT);
        int target = (ticket / NWG + 1) * NWG;   // end of this round
        while (__hip_atomic_load(bar, __ATOMIC_ACQUIRE,
                                 __HIP_MEMORY_SCOPE_AGENT) < target)
            __builtin_amdgcn_s_sleep(2);
    }
    __syncthreads();
    __threadfence();      // acquire for all threads in the block
}

// --------------------------------------------------- fragment assembly -----
// A (16x32 bf16): lane holds M=lane&15; elements 0..7 -> K = base+{0..7},
// elements 8..15 -> K = base+16+{0..7}, base = 8*(lane>=16).
// p points at &state[row][kt*32 + laneBase], 16B-aligned: two 16B runs.
__device__ __forceinline__ v16bf load_a_bf16(const __bf16* __restrict__ p) {
    v16bf a;
#pragma unroll
    for (int e = 0; e < 8; ++e) a[e]     = p[e];
#pragma unroll
    for (int e = 0; e < 8; ++e) a[8 + e] = p[16 + e];
    return a;
}
// B (32x16 bf16): lane holds N=lane&15; same K order as A (base = 8*(lane>=16)).
__device__ __forceinline__ v16bf load_b_lds(const __bf16* __restrict__ p) {
    v16bf b;
#pragma unroll
    for (int e = 0; e < 8; ++e) b[e] = p[e];
#pragma unroll
    for (int e = 0; e < 8; ++e) b[8 + e] = p[16 + e];
    return b;
}

// ---------------------------------------------------------------- main ------
__global__ __launch_bounds__(NTHREADS, 1)
void lmu_main(const float* __restrict__ x,  const float* __restrict__ ie,
              const float* __restrict__ he, const float* __restrict__ me,
              const float* __restrict__ ik, const float* __restrict__ BT,
              const __bf16* __restrict__ hkT, const __bf16* __restrict__ mkT,
              const __bf16* __restrict__ atT,
              float* __restrict__ hbuf, float* __restrict__ mbuf,
              __bf16* __restrict__ hbfbuf, __bf16* __restrict__ mbfbuf,
              int* __restrict__ bar, float* __restrict__ out) {
    __shared__ __bf16 wHK[HCOLS*UNITS];   // [col][k] 32 KB
    __shared__ __bf16 wMK[HCOLS*ORDER];   // [col][k] 16 KB
    __shared__ __bf16 wAT[MCOLS*ORDER];   // [col][k]  8 KB
    __shared__ float  heS[UNITS], meS[ORDER], ikS[HCOLS], btS[MCOLS];
    __shared__ float  uS[BATCH];
    __shared__ float  upart[BATCH][8];

    const int wg   = blockIdx.x;
    const int tid  = threadIdx.x;
    const int wave = tid >> 5, lane = tid & 31;

    // stage loop-invariant weight slices into LDS (linear copies; pre-transposed)
    for (int i = tid; i < HCOLS*UNITS; i += NTHREADS) wHK[i] = hkT[wg*HCOLS*UNITS + i];
    for (int i = tid; i < HCOLS*ORDER; i += NTHREADS) wMK[i] = mkT[wg*HCOLS*ORDER + i];
    for (int i = tid; i < MCOLS*ORDER; i += NTHREADS) wAT[i] = atT[wg*MCOLS*ORDER + i];
    for (int i = tid; i < UNITS; i += NTHREADS) heS[i] = he[i];
    for (int i = tid; i < ORDER; i += NTHREADS) meS[i] = me[i];
    if (tid < HCOLS) ikS[tid] = ik[wg*HCOLS + tid];
    if (tid < MCOLS) btS[tid] = BT[wg*MCOLS + tid];
    const float ie0 = ie[0];
    __syncthreads();

    for (int t = 0; t < TSTEPS; ++t) {
        const float*  hc   = hbuf   + (t & 1)       * (BATCH*UNITS);
        float*        hn   = hbuf   + ((t + 1) & 1) * (BATCH*UNITS);
        const float*  mc   = mbuf   + (t & 1)       * (BATCH*ORDER);
        float*        mn   = mbuf   + ((t + 1) & 1) * (BATCH*ORDER);
        const __bf16* hcb  = hbfbuf + (t & 1)       * (BATCH*UNITS);
        __bf16*       hnb  = hbfbuf + ((t + 1) & 1) * (BATCH*UNITS);
        const __bf16* mcb  = mbfbuf + (t & 1)       * (BATCH*ORDER);
        __bf16*       mnb  = mbfbuf + ((t + 1) & 1) * (BATCH*ORDER);

        // ---- u[b] = x*ie + h@he + m@me (deterministic tree reduction, fp32) --
        {
            int b = tid >> 3, part = tid & 7;
            float s = 0.f;
            const float* hr = hc + b*UNITS + part*64;
            const float* er = heS + part*64;
#pragma unroll 8
            for (int k = 0; k < 64; ++k) s += hr[k]*er[k];
            const float* mr = mc + b*ORDER + part*32;
            const float* fr = meS + part*32;
#pragma unroll 8
            for (int k = 0; k < 32; ++k) s += mr[k]*fr[k];
            upart[b][part] = s;
        }
        __syncthreads();
        if (tid < BATCH) {
            float uu = x[tid*TSTEPS + t] * ie0;
#pragma unroll
            for (int p = 0; p < 8; ++p) uu += upart[tid][p];
            uS[tid] = uu;
        }
        __syncthreads();

        // ---- m_new slice = m + m@AT_slice + u*BT_slice  (waves 0..3) ----
        if (wave < 4) {
            const int mtile = wave;
            const int row = mtile*16 + (lane & 15);
            const int laneBase = (lane >> 4) * 8;
            v8f acc = {};
#pragma unroll
            for (int kt = 0; kt < ORDER/32; ++kt) {
                v16bf a = load_a_bf16(mcb + row*ORDER + kt*32 + laneBase);
                v16bf b = load_b_lds(&wAT[(lane & 15)*ORDER + kt*32 + (lane >> 4)*8]);
                acc = __builtin_amdgcn_wmma_f32_16x16x32_bf16(
                          false, a, false, b, (short)0, acc, false, false);
            }
            const int col16 = lane & 15, half = lane >> 4;
            const int gcol = wg*MCOLS + col16;
#pragma unroll
            for (int r = 0; r < 8; ++r) {
                int rb = mtile*16 + r + 8*half;     // batch row
                float v = acc[r] + mc[rb*ORDER + gcol] + uS[rb]*btS[col16];
                mn [rb*ORDER + gcol] = v;
                mnb[rb*ORDER + gcol] = (__bf16)v;
            }
        }
        grid_barrier(bar);   // m_new (f32 + bf16 mirror) complete everywhere

        // ---- h_new slice = tanh(x*ik + h@hk_slice + m_new@mk_slice) (waves 0..7)
        if (wave < 8) {
            const int mtile = wave >> 1, ntile = wave & 1;
            const int row = mtile*16 + (lane & 15);
            const int laneBase = (lane >> 4) * 8;
            const int colIdx = ntile*16 + (lane & 15);
            v8f acc = {};
#pragma unroll
            for (int kt = 0; kt < UNITS/32; ++kt) {
                v16bf a = load_a_bf16(hcb + row*UNITS + kt*32 + laneBase);
                v16bf b = load_b_lds(&wHK[colIdx*UNITS + kt*32 + (lane >> 4)*8]);
                acc = __builtin_amdgcn_wmma_f32_16x16x32_bf16(
                          false, a, false, b, (short)0, acc, false, false);
            }
#pragma unroll
            for (int kt = 0; kt < ORDER/32; ++kt) {
                v16bf a = load_a_bf16(mnb + row*ORDER + kt*32 + laneBase);
                v16bf b = load_b_lds(&wMK[colIdx*ORDER + kt*32 + (lane >> 4)*8]);
                acc = __builtin_amdgcn_wmma_f32_16x16x32_bf16(
                          false, a, false, b, (short)0, acc, false, false);
            }
            const int col16 = lane & 15, half = lane >> 4;
            const int gcol = wg*HCOLS + ntile*16 + col16;
            const float ikv = ikS[ntile*16 + col16];
#pragma unroll
            for (int r = 0; r < 8; ++r) {
                int rb = mtile*16 + r + 8*half;     // batch row
                float v = tanhf(acc[r] + x[rb*TSTEPS + t]*ikv);
                hn [rb*UNITS + gcol] = v;
                hnb[rb*UNITS + gcol] = (__bf16)v;
                out[(rb*TSTEPS + t)*UNITS + gcol] = v;
            }
        }
        grid_barrier(bar);   // h_new complete everywhere
    }
}

// ---------------------------------------------------------------- launch ----
extern "C" void kernel_launch(void* const* d_in, const int* in_sizes, int n_in,
                              void* d_out, int out_size, void* d_ws, size_t ws_size,
                              hipStream_t stream) {
    const float* x  = (const float*)d_in[0];
    const float* ie = (const float*)d_in[1];
    const float* he = (const float*)d_in[2];
    const float* me = (const float*)d_in[3];
    const float* ik = (const float*)d_in[4];
    const float* hk = (const float*)d_in[5];
    const float* mk = (const float*)d_in[6];
    const float* AT = (const float*)d_in[7];
    const float* BT = (const float*)d_in[8];

    char* ws = (char*)d_ws;
    int*    bar  = (int*)(ws + OFF_BAR);
    float*  hbuf = (float*)(ws + OFF_H);
    float*  mbuf = (float*)(ws + OFF_M);
    __bf16* hkT  = (__bf16*)(ws + OFF_HKT);
    __bf16* mkT  = (__bf16*)(ws + OFF_MKT);
    __bf16* atT  = (__bf16*)(ws + OFF_ATT);
    __bf16* hbf  = (__bf16*)(ws + OFF_HBF);
    __bf16* mbf  = (__bf16*)(ws + OFF_MBF);

    hipLaunchKernelGGL(lmu_prep, dim3(1024), dim3(256), 0, stream,
                       hk, mk, AT, hkT, mkT, atT, hbuf, mbuf, hbf, mbf, bar);
    hipLaunchKernelGGL(lmu_main, dim3(NWG), dim3(NTHREADS), 0, stream,
                       x, ie, he, me, ik, BT, hkT, mkT, atT,
                       hbuf, mbuf, hbf, mbf, bar, (float*)d_out);
}
